// Attention_Embedding_15951508537565
// MI455X (gfx1250) — matrix operand order repaired and checked by reading, not compiled, not run on hardware
//
#include <hip/hip_runtime.h>

typedef __attribute__((ext_vector_type(16))) _Float16 v16h;
typedef __attribute__((ext_vector_type(8)))  _Float16 v8h;
typedef __attribute__((ext_vector_type(8)))  float    v8f;

constexpr int Bc = 4, Hc = 16, Wc = 16, Dc = 16, Cc = 32;
constexpr int Nc  = Hc * Wc * Dc;     // 4096
constexpr int BNC = Bc * Nc * Cc;     // 524288

#define WMMA_F16(a, b, c) \
  __builtin_amdgcn_wmma_f32_16x16x32_f16(false, (a), false, (b), (short)0, (c), false, false)

// A-operand fragment (16xK tile, row-major source, 16-bit elements), contiguous
// K order matching the B loads: lanes 0-15 hold M=lane with K={0..15},
// lanes 16-31 hold M=lane-16 with K={16..31}.
static __device__ __forceinline__ v16h load_A_frag(const _Float16* base, int stride,
                                                   int kc, int lane) {
  int m = lane & 15, g = lane >> 4;
  const _Float16* p = base + m * stride + kc + 16 * g;
  v8h lo = *(const v8h*)p;
  v8h hi = *(const v8h*)(p + 8);
  return __builtin_shufflevector(lo, hi, 0, 1, 2, 3, 4, 5, 6, 7,
                                 8, 9, 10, 11, 12, 13, 14, 15);
}

// ---------------------------------------------------------------------------
// prep: f16 copy of inputs (row-major qh and channel-major qhT) + transposed
// f16 conv weights WT[f][k] with k = t*32 + ci  (k column-contiguous for WMMA B).
// ---------------------------------------------------------------------------
__global__ void prep_kernel(const float* __restrict__ x,
                            const float* __restrict__ w_ch,
                            const float* __restrict__ w_pos,
                            _Float16* __restrict__ qh, _Float16* __restrict__ qhT,
                            _Float16* __restrict__ WchT, _Float16* __restrict__ WposT) {
  int idx = blockIdx.x * blockDim.x + threadIdx.x;
  if (idx < BNC) {
    qh[idx] = (_Float16)x[idx];
    int b   = idx >> 17;              // / (C*N) = 2^17
    int rem = idx & (Cc * Nc - 1);
    int c   = rem >> 12;              // / N (4096)
    int nr  = rem & (Nc - 1);
    qhT[idx] = (_Float16)x[b * Nc * Cc + nr * Cc + c];
  }
  if (idx < 32 * 128) {
    int f = idx >> 7, k = idx & 127;
    int t = k >> 5, ci = k & 31;
    WchT[idx]  = (_Float16)w_ch [t * Cc * Cc + ci * Cc + f];
    WposT[idx] = (_Float16)w_pos[t * Cc * Cc + ci * Cc + f];
  }
}

// ---------------------------------------------------------------------------
// Flash-style channel attention, split-K over 2 waves per block:
//   caH = f16( beta * softmax(q qT) q + x )
// Each wave owns a 16-row query tile and half the 4096 keys (64 chunks of 32);
// partial (m, l, O) states are merged through LDS at the end.
// ---------------------------------------------------------------------------
__global__ __launch_bounds__(64) void flash_channel_attn(
    const _Float16* __restrict__ qh, const _Float16* __restrict__ qhT,
    const float* __restrict__ x, const float* __restrict__ beta,
    _Float16* __restrict__ caH) {
  __shared__ __align__(32) float    sS[2 * 512];     // per-wave S tile (16x32 f32)
  __shared__ __align__(32) _Float16 sP[2 * 512];     // per-wave P tile (16x32 f16)
  __shared__ __align__(32) float    sScale[2 * 16];
  __shared__ __align__(32) float    sM[32], sL[32];  // per-wave row stats
  __shared__ __align__(32) float    sAcc[2 * 512];   // per-wave O accumulators

  int tid  = threadIdx.x;
  int wid  = tid >> 5;                  // split-K wave id (0/1)
  int lane = tid & 31;
  int b    = blockIdx.x >> 8;           // 256 tiles per batch
  int row0 = (blockIdx.x & 255) << 4;
  int n = lane & 15, g = lane >> 4, m = n;

  const _Float16* qh_b  = qh  + b * Nc * Cc;
  const _Float16* qhT_b = qhT + b * Cc * Nc;
  float*    sSw = sS + wid * 512;
  _Float16* sPw = sP + wid * 512;
  float*    sCw = sScale + wid * 16;

  v16h aQ = load_A_frag(qh_b + row0 * Cc, Cc, 0, lane);
  float m_i = -3.0e38f, l_i = 0.0f;
  v8f acc0 = {}, acc1 = {};

  int kb0 = wid * (Nc / 2);
  for (int kb = kb0; kb < kb0 + Nc / 2; kb += 32) {
    __builtin_prefetch(qh_b + (kb + 32) * Cc, 0, 1);

    // S = Q * K^T : B-operand columns are key rows -> contiguous 32B per lane.
    v16h b0 = *(const v16h*)(qh_b + (kb + n) * Cc + 16 * g);
    v16h b1 = *(const v16h*)(qh_b + (kb + 16 + n) * Cc + 16 * g);
    // V B-fragments issued early so their latency hides under the softmax.
    v16h bv0 = *(const v16h*)(qhT_b + n * Nc + kb + 16 * g);
    v16h bv1 = *(const v16h*)(qhT_b + (16 + n) * Nc + kb + 16 * g);

    v8f z = {};
    v8f s0 = WMMA_F16(aQ, b0, z);
    v8f s1 = WMMA_F16(aQ, b1, z);

    // C-layout -> LDS row-major S tile (16 x 32)
#pragma unroll
    for (int r = 0; r < 8; ++r) {
      sSw[(r + 8 * g) * 32 + n]      = s0[r];
      sSw[(r + 8 * g) * 32 + n + 16] = s1[r];
    }
    __syncthreads();

    // online softmax: lane pair (m, m+16) owns row m, each handles 16 cols
    float vals[16];
    float vmax = -3.0e38f;
#pragma unroll
    for (int i = 0; i < 16; ++i) {
      vals[i] = sSw[m * 32 + 16 * g + i];
      vmax = fmaxf(vmax, vals[i]);
    }
    vmax = fmaxf(vmax, __shfl_xor(vmax, 16));
    float mnew    = fmaxf(m_i, vmax);
    float rescale = __expf(m_i - mnew);
    float lsum = 0.f;
#pragma unroll
    for (int i = 0; i < 16; ++i) {
      float p = __expf(vals[i] - mnew);
      lsum += p;
      sPw[m * 32 + 16 * g + i] = (_Float16)p;
    }
    lsum += __shfl_xor(lsum, 16);
    l_i = l_i * rescale + lsum;
    m_i = mnew;
    if (g == 0) sCw[m] = rescale;
    __syncthreads();

    // rescale O accumulators per row (C-layout: vgpr r <-> row r + 8g)
#pragma unroll
    for (int r = 0; r < 8; ++r) {
      float sc = sCw[r + 8 * g];
      acc0[r] *= sc;
      acc1[r] *= sc;
    }

    // O += P * V : P re-read from LDS in A-layout; V from channel-major qhT
    v16h aP = load_A_frag(sPw, 32, 0, lane);
    acc0 = WMMA_F16(aP, bv0, acc0);
    acc1 = WMMA_F16(aP, bv1, acc1);
    __syncthreads();
  }

  // publish per-wave partial state
  if (g == 0) { sM[wid * 16 + m] = m_i; sL[wid * 16 + m] = l_i; }
#pragma unroll
  for (int r = 0; r < 8; ++r) {
    sAcc[wid * 512 + (r + 8 * g) * 32 + n]      = acc0[r];
    sAcc[wid * 512 + (r + 8 * g) * 32 + n + 16] = acc1[r];
  }
  __syncthreads();

  // wave 0 merges the two split-K states and writes the residual-added f16 tile
  if (wid == 0) {
    float bet = beta[0];
    const float* x_b  = x   + b * Nc * Cc;
    _Float16*   ca_b  = caH + b * Nc * Cc;
#pragma unroll
    for (int r = 0; r < 8; ++r) {
      int mr = r + 8 * g;
      float m0 = sM[mr], m1 = sM[16 + mr];
      float mt = fmaxf(m0, m1);
      float e0 = __expf(m0 - mt), e1 = __expf(m1 - mt);
      float inv = 1.0f / (sL[mr] * e0 + sL[16 + mr] * e1);
      int row = row0 + mr;
#pragma unroll
      for (int h = 0; h < 2; ++h) {
        int c = n + 16 * h;
        float o = (sAcc[mr * 32 + c] * e0 + sAcc[512 + mr * 32 + c] * e1) * inv;
        ca_b[row * Cc + c] = (_Float16)(bet * o + x_b[row * Cc + c]);
      }
    }
  }
}

// ---------------------------------------------------------------------------
// Position branch (0.2 GFLOP total -> plain VALU f32)
// ---------------------------------------------------------------------------
__global__ void proj_qkv(const float* __restrict__ x,
                         const float* __restrict__ wq, const float* __restrict__ bq,
                         const float* __restrict__ wk, const float* __restrict__ bk,
                         const float* __restrict__ wv, const float* __restrict__ bv,
                         float* __restrict__ Qm, float* __restrict__ Km,
                         float* __restrict__ Vm) {
  int idx = blockIdx.x * blockDim.x + threadIdx.x;
  if (idx >= BNC) return;
  int f = idx & 31;
  const float* xr = x + (idx & ~31);
  float q = bq[f], k = bk[f], v = bv[f];
#pragma unroll 8
  for (int c = 0; c < 32; ++c) {
    float xv = xr[c];
    q = fmaf(xv, wq[c * 32 + f], q);
    k = fmaf(xv, wk[c * 32 + f], k);
    v = fmaf(xv, wv[c * 32 + f], v);
  }
  Qm[idx] = q; Km[idx] = k; Vm[idx] = v;
}

__global__ __launch_bounds__(1024) void chan_energy(
    const float* __restrict__ Qm, const float* __restrict__ Km,
    float* __restrict__ attn2) {
  __shared__ float sQ[4096];
  __shared__ float sK[4096];
  int t = threadIdx.x, b = blockIdx.x;
  int c = t >> 5, f = t & 31;
  const float* Qb = Qm + b * Nc * Cc;
  const float* Kb = Km + b * Nc * Cc;
  float acc = 0.f;
  for (int ch = 0; ch < Nc * Cc; ch += 4096) {
    for (int i = t; i < 4096; i += 1024) { sQ[i] = Qb[ch + i]; sK[i] = Kb[ch + i]; }
    __syncthreads();
#pragma unroll 8
    for (int r = 0; r < 128; ++r)
      acc = fmaf(sQ[r * 32 + c], sK[r * 32 + f], acc);
    __syncthreads();
  }
  // softmax over f (lanes of this wave32)
  float mx = acc;
  for (int off = 16; off; off >>= 1) mx = fmaxf(mx, __shfl_xor(mx, off));
  float p = __expf(acc - mx);
  float sm = p;
  for (int off = 16; off; off >>= 1) sm += __shfl_xor(sm, off);
  attn2[b * Cc * Cc + c * Cc + f] = p / sm;
}

__global__ void pos_apply(const float* __restrict__ Vm, const float* __restrict__ attn2,
                          const float* __restrict__ x, const float* __restrict__ gamma,
                          _Float16* __restrict__ paH) {
  int idx = blockIdx.x * blockDim.x + threadIdx.x;
  if (idx >= BNC) return;
  int b = idx >> 17;
  int f = idx & 31;
  const float* vr = Vm + (idx & ~31);
  const float* ar = attn2 + b * Cc * Cc + f * Cc;   // attn2[b, f, c]
  float s = 0.f;
#pragma unroll 8
  for (int c = 0; c < 32; ++c) s = fmaf(vr[c], ar[c], s);
  paH[idx] = (_Float16)(gamma[0] * s + x[idx]);
}

// ---------------------------------------------------------------------------
// Conv3D(1,1,4)+ReLU on both branches, summed. Sliding-window GEMM
// (M=16 depth rows per (b,h,w), K=128, N=32) on WMMA f16, 13 valid rows stored.
// A-fragments come straight from the f16 branch buffers (L2-resident) -- no
// LDS staging needed; windows m>=13 read tail-padded garbage and are masked.
// ---------------------------------------------------------------------------
__global__ __launch_bounds__(32) void conv_embed(
    const _Float16* __restrict__ caH, const _Float16* __restrict__ paH,
    const _Float16* __restrict__ WchT, const _Float16* __restrict__ WposT,
    const float* __restrict__ b_ch, const float* __restrict__ b_pos,
    float* __restrict__ out) {
  int lane = threadIdx.x;
  int bhw  = blockIdx.x;
  const _Float16* xc = caH + bhw * Dc * Cc;
  const _Float16* xp = paH + bhw * Dc * Cc;

  int n = lane & 15, g = lane >> 4;
  v8f c0 = {}, c1 = {}, p0 = {}, p1 = {};
#pragma unroll
  for (int kc = 0; kc < 128; kc += 32) {
    v16h axc = load_A_frag(xc, Cc, kc, lane);   // A[m=d][k] = X[d*32 + k]
    v16h axp = load_A_frag(xp, Cc, kc, lane);
    v16h w0 = *(const v16h*)(WchT  + n        * 128 + kc + 16 * g);
    v16h w1 = *(const v16h*)(WchT  + (16 + n) * 128 + kc + 16 * g);
    v16h u0 = *(const v16h*)(WposT + n        * 128 + kc + 16 * g);
    v16h u1 = *(const v16h*)(WposT + (16 + n) * 128 + kc + 16 * g);
    c0 = WMMA_F16(axc, w0, c0);
    c1 = WMMA_F16(axc, w1, c1);
    p0 = WMMA_F16(axp, u0, p0);
    p1 = WMMA_F16(axp, u1, p1);
  }

  float bc0 = b_ch[n],  bc1 = b_ch[16 + n];
  float bq0 = b_pos[n], bq1 = b_pos[16 + n];
  float* ob = out + bhw * 13 * Cc;
#pragma unroll
  for (int r = 0; r < 8; ++r) {
    int m = r + 8 * g;
    if (m < 13) {
      ob[m * Cc + n]      = fmaxf(c0[r] + bc0, 0.f) + fmaxf(p0[r] + bq0, 0.f);
      ob[m * Cc + n + 16] = fmaxf(c1[r] + bc1, 0.f) + fmaxf(p1[r] + bq1, 0.f);
    }
  }
}

// ---------------------------------------------------------------------------
extern "C" void kernel_launch(void* const* d_in, const int* in_sizes, int n_in,
                              void* d_out, int out_size, void* d_ws, size_t ws_size,
                              hipStream_t stream) {
  (void)in_sizes; (void)n_in; (void)out_size; (void)ws_size;
  const float* x     = (const float*)d_in[0];
  const float* beta  = (const float*)d_in[1];
  const float* wq    = (const float*)d_in[2];
  const float* bq    = (const float*)d_in[3];
  const float* wk    = (const float*)d_in[4];
  const float* bk    = (const float*)d_in[5];
  const float* wv    = (const float*)d_in[6];
  const float* bv    = (const float*)d_in[7];
  const float* gamma = (const float*)d_in[8];
  const float* w_ch  = (const float*)d_in[9];
  const float* b_ch  = (const float*)d_in[10];
  const float* w_pos = (const float*)d_in[11];
  const float* b_pos = (const float*)d_in[12];
  float* out = (float*)d_out;

  char* w = (char*)d_ws;                           // all offsets 256B aligned
  _Float16* qh    = (_Float16*)(w + 0);            // 1,048,576 B
  _Float16* qhT   = (_Float16*)(w + 1048576);      // 1,048,576 B
  _Float16* WchT  = (_Float16*)(w + 2097152);      // 8,192 B
  _Float16* WposT = (_Float16*)(w + 2105344);      // 8,192 B
  _Float16* caH   = (_Float16*)(w + 2113536);      // (BNC+256)*2 = 1,049,088 B
  _Float16* paH   = (_Float16*)(w + 3162624);      // 1,049,088 B
  float*    Qm    = (float*)  (w + 4211712);       // 2,097,152 B
  float*    Km    = (float*)  (w + 6308864);       // 2,097,152 B
  float*    Vm    = (float*)  (w + 8406016);       // 2,097,152 B
  float*    attn2 = (float*)  (w + 10503168);      // 16,384 B  (total ~10.5 MB)

  prep_kernel<<<(BNC + 255) / 256, 256, 0, stream>>>(x, w_ch, w_pos, qh, qhT, WchT, WposT);
  proj_qkv<<<(BNC + 255) / 256, 256, 0, stream>>>(x, wq, bq, wk, bk, wv, bv, Qm, Km, Vm);
  chan_energy<<<Bc, 1024, 0, stream>>>(Qm, Km, attn2);
  flash_channel_attn<<<Bc * (Nc / 16), 64, 0, stream>>>(qh, qhT, x, beta, caH);
  pos_apply<<<(BNC + 255) / 256, 256, 0, stream>>>(Vm, attn2, x, gamma, paH);
  conv_embed<<<Bc * Hc * Wc, 32, 0, stream>>>(caH, paH, WchT, WposT, b_ch, b_pos, out);
}
